// LVCBlock_20890720928161
// MI455X (gfx1250) — matrix operand order repaired and checked by reading, not compiled, not run on hardware
//
#include <hip/hip_runtime.h>
#include <cmath>

typedef __attribute__((ext_vector_type(16))) __bf16 v16bf;
typedef __attribute__((ext_vector_type(8)))  __bf16 v8bf;
typedef __attribute__((ext_vector_type(8)))  float  v8f;

#define DEVFN static __device__ __forceinline__

namespace lvc {
constexpr int NB = 32;    // batch
constexpr int C  = 512;   // channels
constexpr int CM = 128;   // mid channels
constexpr int KC = 64;    // codewords
constexpr int HW = 1024;  // 32x32 pixels
constexpr int WPIX = 32;

constexpr int BM = 64;    // block tile M (4 wave rows)
constexpr int BN = 64;    // block tile N (2 wave cols x 32)
constexpr int BK = 32;    // k step
constexpr int LDT = 40;   // padded LDS row stride in bf16 (80B: 16B aligned)
}

DEVFN int imin(int a, int b) { return a < b ? a : b; }

DEVFN __bf16 f2bf(float f) {
  unsigned u = __builtin_bit_cast(unsigned, f);
  unsigned r = u + 0x7FFFu + ((u >> 16) & 1u);   // round-to-nearest-even
  unsigned short h = (unsigned short)(r >> 16);
  return __builtin_bit_cast(__bf16, h);
}
DEVFN float bf2f(__bf16 b) {
  unsigned short s = __builtin_bit_cast(unsigned short, b);
  unsigned u = ((unsigned)s) << 16;
  return __builtin_bit_cast(float, u);
}

DEVFN v8f wmma_bf16(const v16bf& a, const v16bf& b, const v8f& c) {
  return __builtin_amdgcn_wmma_f32_16x16x32_bf16(false, a, false, b,
                                                 (short)0, c, false, false);
}
DEVFN v16bf join16(v8bf lo, v8bf hi) {
  v16bf r;
  #pragma unroll
  for (int i = 0; i < 8; ++i) { r[i] = lo[i]; r[i + 8] = hi[i]; }
  return r;
}

// ---------------------------------------------------------------------------
// Generic strided batched GEMM:  C[z] = act( A[z]*B[z]*scale + bias (+resid) )
// Template: AK1 -> sAk==1 (else requires sAm==1); BK1 -> sBk==1 (else sBn==1).
// Requirements: Kd % 32 == 0; ragged M/N handled by clamp + masked stores.
// ---------------------------------------------------------------------------
struct GemmArgs {
  const __bf16* A;  int sAm, sAk; long long bA;
  const __bf16* Bm; int sBk, sBn; long long bB;
  float*  Cf; __bf16* Cb; int sCm, sCn; long long bC;
  const float* resid;   // optional fp32 add, same strides as C
  const float* scale;   // per-row (per-col if perN) multiplier, or null
  const float* bias;    // per-row (per-col if perN) add, or null
  float bnInv;          // folded 1/sqrt(1+eps)
  int M, N, Kd;
  int act;              // 0 none, 1 relu, 2 sigmoid
  int perN;
};

template <bool AK1, bool BK1>
__global__ __launch_bounds__(256) void gemm_bf16_wmma(GemmArgs g)
{
  using namespace lvc;
  __shared__ alignas(16) __bf16 Al[BM * LDT];
  __shared__ alignas(16) __bf16 Bt[BN * LDT];   // transposed: [n][k]

  const int t    = threadIdx.x;
  const int wave = t >> 5, lane = t & 31;
  const int wm   = wave >> 1, wn = wave & 1;    // 4x2 waves, each 16(M)x32(N)
  const int half = lane >> 4, r16 = lane & 15;
  const int m0   = blockIdx.y * BM;
  const int n0   = blockIdx.x * BN;
  const long long z = blockIdx.z;

  const __bf16* Ab = g.A  + z * g.bA;
  const __bf16* Bb = g.Bm + z * g.bB;

  // -------- staging setup (one 16B vector load per thread per tile) --------
  const __bf16* apK;
  __bf16* asl;
  int aStep;
  if constexpr (AK1) {          // thread: row = t>>2, kseg = (t&3)*8 contig k
    const int arow = imin(m0 + (t >> 2), g.M - 1);
    apK   = Ab + arow * g.sAm + (t & 3) * 8;
    asl   = &Al[(t >> 2) * LDT + (t & 3) * 8];
    aStep = BK;
  } else {                      // sAm==1: thread: k = t>>3, 8 contig m
    const int am0 = imin(m0 + (t & 7) * 8, g.M - 8);
    apK   = Ab + (t >> 3) * g.sAk + am0;
    asl   = &Al[((t & 7) * 8) * LDT + (t >> 3)];
    aStep = BK * g.sAk;
  }
  const __bf16* bpK;
  __bf16* bsl;
  int bStep;
  if constexpr (BK1) {          // thread: n = t>>2, kseg = (t&3)*8 contig k
    const int bn = imin(n0 + (t >> 2), g.N - 1);
    bpK   = Bb + bn * g.sBn + (t & 3) * 8;
    bsl   = &Bt[(t >> 2) * LDT + (t & 3) * 8];
    bStep = BK;
  } else {                      // sBn==1: thread: k = t>>3, 8 contig n
    const int bn0 = imin(n0 + (t & 7) * 8, g.N - 8);
    bpK   = Bb + (t >> 3) * g.sBk + bn0;
    bsl   = &Bt[((t & 7) * 8) * LDT + (t >> 3)];
    bStep = BK * g.sBk;
  }

  v8f acc0 = {}, acc1 = {};
  for (int k0 = 0; k0 < g.Kd; k0 += BK) {
    const v8bf av = *(const v8bf*)apK;
    const v8bf bv = *(const v8bf*)bpK;
    if constexpr (AK1) {
      *(v8bf*)asl = av;
    } else {
      #pragma unroll
      for (int j = 0; j < 8; ++j) asl[j * LDT] = av[j];
    }
    if constexpr (BK1) {
      *(v8bf*)bsl = bv;
    } else {
      #pragma unroll
      for (int j = 0; j < 8; ++j) bsl[j * LDT] = bv[j];
    }
    __syncthreads();

    if (k0 + BK < g.Kd) {        // prefetch next tiles into cache
      __builtin_prefetch(apK + aStep, 0, 1);
      __builtin_prefetch(bpK + bStep, 0, 1);
    }

    // A 16x32 frag: lanes<16 -> k {0..7,16..23}; lanes>=16 -> {8..15,24..31}
    const __bf16* ap = &Al[(wm * 16 + r16) * LDT + half * 8];
    v16bf af = join16(*(const v8bf*)(ap), *(const v8bf*)(ap + 16));
    // B 32x16 frags: lane col = r16, lanes<16 -> k {0..7,16..23}; lanes>=16 -> {8..15,24..31}
    const __bf16* bp0 = &Bt[(wn * 32 + r16) * LDT + half * 8];
    const __bf16* bp1 = bp0 + 16 * LDT;
    v16bf bf0 = join16(*(const v8bf*)(bp0), *(const v8bf*)(bp0 + 16));
    v16bf bf1 = join16(*(const v8bf*)(bp1), *(const v8bf*)(bp1 + 16));

    acc0 = wmma_bf16(af, bf0, acc0);
    acc1 = wmma_bf16(af, bf1, acc1);
    __syncthreads();

    apK += aStep; bpK += bStep;
  }

  // epilogue: C/D layout: vgpr rr -> M = rr + half*8, N = lane&15
  float*       Cfb = g.Cf    ? g.Cf    + z * g.bC : nullptr;
  __bf16*      Cbb = g.Cb    ? g.Cb    + z * g.bC : nullptr;
  const float* Rb  = g.resid ? g.resid + z * g.bC : nullptr;
  const int gnb = n0 + wn * 32 + r16;
  #pragma unroll
  for (int pp = 0; pp < 2; ++pp) {
    const int gn = gnb + pp * 16;
    const v8f acc = pp ? acc1 : acc0;
    if (gn < g.N) {
      #pragma unroll
      for (int rr = 0; rr < 8; ++rr) {
        const int gm = m0 + wm * 16 + half * 8 + rr;
        if (gm < g.M) {
          float v = acc[rr];
          const int si = g.perN ? gn : gm;
          if (g.scale) v *= g.scale[si] * g.bnInv;
          if (g.bias)  v += g.bias[si];
          const int co = gm * g.sCm + gn * g.sCn;
          if (Rb) v += Rb[co];
          if (g.act == 1)      v = v > 0.f ? v : 0.f;
          else if (g.act == 2) v = 1.f / (1.f + __expf(-v));
          if (Cfb) Cfb[co] = v;
          if (Cbb) Cbb[co] = f2bf(v);
        }
      }
    }
  }
}

// ---------------------------------------------------------------------------
// 3x3 conv as implicit GEMM: M=CM, N=HW, Ktot=9*CM, weights [tap][co][ci]
// ---------------------------------------------------------------------------
__global__ __launch_bounds__(256) void conv3_bf16_wmma(
    const __bf16* __restrict__ w2b, const __bf16* __restrict__ h1,
    __bf16* __restrict__ h2,
    const float* __restrict__ bn_g, const float* __restrict__ bn_b, float bnInv)
{
  using namespace lvc;
  __shared__ alignas(16) __bf16 Al[BM * LDT];
  __shared__ alignas(16) __bf16 Bt[BN * LDT];

  const int t    = threadIdx.x;
  const int wave = t >> 5, lane = t & 31;
  const int wm   = wave >> 1, wn = wave & 1;
  const int half = lane >> 4, r16 = lane & 15;
  const int m0   = blockIdx.y * BM;
  const int n0   = blockIdx.x * BN;
  const long long z = blockIdx.z;
  const __bf16* hb = h1 + z * (long long)(CM * HW);

  // A staging: row = t>>2 (64 rows), kseg = (t&3)*8 (ci contiguous)
  const int arow = t >> 2, akseg = (t & 3) * 8;
  __bf16* asl = &Al[arow * LDT + akseg];
  // B staging: k = t>>3 (32 ci), 8 contiguous pixels at (t&7)*8
  const int bk = t >> 3, nseg = (t & 7) * 8;
  const int n1 = n0 + nseg;
  const int prow = n1 >> 5, pcol0 = n1 & 31;   // same image row for all 8
  __bf16* bsl = &Bt[nseg * LDT + bk];

  v8f acc0 = {}, acc1 = {};
  for (int k0 = 0; k0 < 9 * CM; k0 += BK) {   // tap constant per k-step
    const int tap = k0 >> 7;
    const int ci0 = k0 & (CM - 1);
    const int dh = tap / 3 - 1, dw = tap % 3 - 1;

    // A: one b128 load + one b128 LDS store
    const __bf16* wp = w2b + tap * CM * CM + (m0 + arow) * CM + ci0 + akseg;
    *(v8bf*)asl = *(const v8bf*)wp;

    // B: load 8 masked pixels into regs, then store (batched waits)
    const int hh = prow + dh;
    const bool rowok = (hh >= 0) & (hh < WPIX);
    const __bf16* bp = hb + (ci0 + bk) * HW + (rowok ? hh * WPIX : 0);
    unsigned short tmp[8];
    #pragma unroll
    for (int j = 0; j < 8; ++j) {
      const int ww = pcol0 + dw + j;
      const bool ok = rowok & (ww >= 0) & (ww < WPIX);
      unsigned short u = __builtin_bit_cast(unsigned short, bp[ok ? ww : 0]);
      tmp[j] = ok ? u : (unsigned short)0;
    }
    #pragma unroll
    for (int j = 0; j < 8; ++j)
      bsl[j * LDT] = __builtin_bit_cast(__bf16, tmp[j]);
    __syncthreads();

    const __bf16* ap = &Al[(wm * 16 + r16) * LDT + half * 8];
    v16bf af = join16(*(const v8bf*)(ap), *(const v8bf*)(ap + 16));
    const __bf16* bp0 = &Bt[(wn * 32 + r16) * LDT + half * 8];
    const __bf16* bp1 = bp0 + 16 * LDT;
    v16bf bf0 = join16(*(const v8bf*)(bp0), *(const v8bf*)(bp0 + 16));
    v16bf bf1 = join16(*(const v8bf*)(bp1), *(const v8bf*)(bp1 + 16));
    acc0 = wmma_bf16(af, bf0, acc0);
    acc1 = wmma_bf16(af, bf1, acc1);
    __syncthreads();
  }

  __bf16* outp = h2 + z * (long long)(CM * HW);
  const int gnb = n0 + wn * 32 + r16;
  #pragma unroll
  for (int pp = 0; pp < 2; ++pp) {
    const int gn = gnb + pp * 16;
    const v8f acc = pp ? acc1 : acc0;
    #pragma unroll
    for (int rr = 0; rr < 8; ++rr) {
      const int gm = m0 + wm * 16 + half * 8 + rr;
      float v = acc[rr] * (bn_g[gm] * bnInv) + bn_b[gm];
      v = v > 0.f ? v : 0.f;
      outp[gm * HW + gn] = f2bf(v);
    }
  }
}

// ---------------------------------------------------------------------------
// small elementwise / reduction kernels
// ---------------------------------------------------------------------------
__global__ void cvt_bf16_kernel(const float* __restrict__ in,
                                __bf16* __restrict__ out, int n) {
  int i = blockIdx.x * 256 + threadIdx.x;
  if (i < n) out[i] = f2bf(in[i]);
}

__global__ void cvt_w2_kernel(const float* __restrict__ in,
                              __bf16* __restrict__ out) {  // [co][ci][3][3]->[tap][co][ci]
  using namespace lvc;
  int i = blockIdx.x * 256 + threadIdx.x;
  if (i >= CM * CM * 9) return;
  int co = i / (CM * 9);
  int rem = i % (CM * 9);
  int ci = rem / 9, tp = rem % 9;
  out[tp * CM * CM + co * CM + ci] = f2bf(in[i]);
}

__global__ void x2_kernel(const __bf16* __restrict__ e, float* __restrict__ x2) {
  using namespace lvc;
  int n = blockIdx.x * 256 + threadIdx.x;
  int b = blockIdx.y;
  const __bf16* eb = e + (long long)b * C * HW + n;
  float s = 0.f;
  for (int c = 0; c < C; ++c) { float v = bf2f(eb[c * HW]); s += v * v; }
  x2[b * HW + n] = s;
}

__global__ void c2_kernel(const float* __restrict__ cw, float* __restrict__ c2n) {
  using namespace lvc;
  int k = threadIdx.x;
  if (k >= KC) return;
  float s = 0.f;
  for (int c = 0; c < C; ++c) { float v = cw[k * C + c]; s += v * v; }
  c2n[k] = s;
}

__global__ void softmax_kernel(const float* __restrict__ xc, const float* __restrict__ x2,
                               const float* __restrict__ c2n, const float* __restrict__ sc,
                               __bf16* __restrict__ a) {
  using namespace lvc;
  int n = blockIdx.x * 256 + threadIdx.x;
  int b = blockIdx.y;
  if (n >= HW) return;
  const float* row = xc + ((long long)b * HW + n) * KC;
  float xx = x2[b * HW + n];
  float mx = -3.0e38f;
  for (int k = 0; k < KC; ++k) {
    float s = sc[k] * (xx - 2.f * row[k] + c2n[k]);
    mx = s > mx ? s : mx;
  }
  float sum = 0.f;
  for (int k = 0; k < KC; ++k)
    sum += __expf(sc[k] * (xx - 2.f * row[k] + c2n[k]) - mx);
  float inv = 1.f / sum;
  __bf16* ar = a + ((long long)b * HW + n) * KC;
  for (int k = 0; k < KC; ++k)
    ar[k] = f2bf(__expf(sc[k] * (xx - 2.f * row[k] + c2n[k]) - mx) * inv);
}

__global__ void asum_kernel(const __bf16* __restrict__ a, float* __restrict__ asum) {
  using namespace lvc;
  int i = blockIdx.x * 256 + threadIdx.x;     // b*KC + k
  if (i >= NB * KC) return;
  int b = i >> 6, k = i & (KC - 1);
  const __bf16* ab = a + (long long)b * HW * KC + k;
  float s = 0.f;
  for (int n = 0; n < HW; ++n) s += bf2f(ab[n * KC]);
  asum[i] = s;
}

__global__ void encpost_kernel(const float* __restrict__ part1, const float* __restrict__ asum,
                               const float* __restrict__ cw, const float* __restrict__ g,
                               const float* __restrict__ bb, float inv,
                               __bf16* __restrict__ enb) {
  using namespace lvc;
  int i = blockIdx.x * 256 + threadIdx.x;     // b*C + c
  if (i >= NB * C) return;
  int b = i >> 9, c = i & (C - 1);
  const float* p  = part1 + (long long)b * KC * C + c;
  const float* as = asum + b * KC;
  float s = 0.f;
  for (int k = 0; k < KC; ++k) {
    float v = (p[k * C] - as[k] * cw[k * C + c]) * (g[k] * inv) + bb[k];
    s += v > 0.f ? v : 0.f;
  }
  enb[i] = f2bf(s * (1.f / KC));
}

__global__ void final_kernel(const float* __restrict__ xb, const float* __restrict__ gam,
                             float* __restrict__ out) {
  using namespace lvc;
  long long i = (long long)blockIdx.x * 256 + threadIdx.x;
  if (i >= (long long)NB * C * HW) return;
  long long bc = i >> 10;                      // (b*C + c)
  float v = xb[i] * (1.f + gam[bc]);
  out[i] = v > 0.f ? v : 0.f;
}

// ---------------------------------------------------------------------------
extern "C" void kernel_launch(void* const* d_in, const int* in_sizes, int n_in,
                              void* d_out, int out_size, void* d_ws, size_t ws_size,
                              hipStream_t stream)
{
  using namespace lvc;
  (void)in_sizes; (void)n_in; (void)out_size; (void)ws_size;

  const float* x       = (const float*)d_in[0];
  const float* c1_w    = (const float*)d_in[1];
  const float* bn1_g   = (const float*)d_in[2];
  const float* bn1_b   = (const float*)d_in[3];
  const float* c2_w    = (const float*)d_in[4];
  const float* bn2_g   = (const float*)d_in[5];
  const float* bn2_b   = (const float*)d_in[6];
  const float* c3_w    = (const float*)d_in[7];
  const float* bn3_g   = (const float*)d_in[8];
  const float* bn3_b   = (const float*)d_in[9];
  const float* rc_w    = (const float*)d_in[10];
  const float* rbn_g   = (const float*)d_in[11];
  const float* rbn_b   = (const float*)d_in[12];
  const float* lvc_w   = (const float*)d_in[13];
  const float* lvcbn_g = (const float*)d_in[14];
  const float* lvcbn_b = (const float*)d_in[15];
  const float* cw      = (const float*)d_in[16];
  const float* escale  = (const float*)d_in[17];
  const float* enbn_g  = (const float*)d_in[18];
  const float* enbn_b  = (const float*)d_in[19];
  const float* fc_w    = (const float*)d_in[20];
  const float* fc_b    = (const float*)d_in[21];
  float* out = (float*)d_out;

  char* base = (char*)d_ws;
  size_t off = 0;
  auto alloc = [&](size_t bytes) -> void* {
    void* p = base + off;
    off = (off + bytes + 255) & ~(size_t)255;
    return p;
  };
  __bf16* xb16  = (__bf16*)alloc((size_t)NB * C * HW * 2);   // bf16(x)
  __bf16* h1    = (__bf16*)alloc((size_t)NB * CM * HW * 2);
  __bf16* h2    = (__bf16*)alloc((size_t)NB * CM * HW * 2);
  float*  rxb   = (float*) alloc((size_t)NB * C * HW * 4);   // r, then xb in place
  __bf16* xbb   = (__bf16*)alloc((size_t)NB * C * HW * 2);   // bf16(xb)
  __bf16* ebuf  = (__bf16*)alloc((size_t)NB * C * HW * 2);   // bf16(e)
  float*  xcb   = (float*) alloc((size_t)NB * HW * KC * 4);
  float*  x2b   = (float*) alloc((size_t)NB * HW * 4);
  __bf16* abuf  = (__bf16*)alloc((size_t)NB * HW * KC * 2);
  float*  asumb = (float*) alloc((size_t)NB * KC * 4);
  float*  part1 = (float*) alloc((size_t)NB * KC * C * 4);
  __bf16* enb   = (__bf16*)alloc((size_t)NB * C * 2);
  float*  gam   = (float*) alloc((size_t)NB * C * 4);
  float*  c2n   = (float*) alloc((size_t)KC * 4);
  __bf16* c1wb  = (__bf16*)alloc((size_t)CM * C * 2);
  __bf16* w2b   = (__bf16*)alloc((size_t)9 * CM * CM * 2);
  __bf16* c3wb  = (__bf16*)alloc((size_t)C * CM * 2);
  __bf16* rcwb  = (__bf16*)alloc((size_t)C * C * 2);
  __bf16* lvcwb = (__bf16*)alloc((size_t)C * C * 2);
  __bf16* cwb   = (__bf16*)alloc((size_t)KC * C * 2);
  __bf16* fcwb  = (__bf16*)alloc((size_t)C * C * 2);

  const float inv_cb  = 1.0f / sqrtf(1.0f + 1e-6f);
  const float inv_lvc = 1.0f / sqrtf(1.0f + 1e-5f);

  auto cvt = [&](const float* src, __bf16* dst, int n) {
    cvt_bf16_kernel<<<(n + 255) / 256, 256, 0, stream>>>(src, dst, n);
  };
  cvt(x, xb16, NB * C * HW);
  cvt(c1_w, c1wb, CM * C);
  cvt(c3_w, c3wb, C * CM);
  cvt(rc_w, rcwb, C * C);
  cvt(lvc_w, lvcwb, C * C);
  cvt(cw, cwb, KC * C);
  cvt(fc_w, fcwb, C * C);
  cvt_w2_kernel<<<(CM * CM * 9 + 255) / 256, 256, 0, stream>>>(c2_w, w2b);

  auto run_gemm = [&](GemmArgs g, int z) {
    dim3 grid((g.N + BN - 1) / BN, (g.M + BM - 1) / BM, z);
    if (g.sAk == 1 && g.sBk == 1)
      gemm_bf16_wmma<true, true><<<grid, 256, 0, stream>>>(g);
    else if (g.sAk == 1)
      gemm_bf16_wmma<true, false><<<grid, 256, 0, stream>>>(g);
    else if (g.sBk == 1)
      gemm_bf16_wmma<false, true><<<grid, 256, 0, stream>>>(g);
    else
      gemm_bf16_wmma<false, false><<<grid, 256, 0, stream>>>(g);
  };

  // c1: h1 = relu(bn1(W1 @ x))     M=CM, N=HW, K=C
  { GemmArgs g = {};
    g.A = c1wb; g.sAm = C; g.sAk = 1; g.bA = 0;
    g.Bm = xb16; g.sBk = HW; g.sBn = 1; g.bB = (long long)C * HW;
    g.Cf = nullptr; g.Cb = h1; g.sCm = HW; g.sCn = 1; g.bC = (long long)CM * HW;
    g.resid = nullptr; g.scale = bn1_g; g.bias = bn1_b; g.bnInv = inv_cb;
    g.M = CM; g.N = HW; g.Kd = C; g.act = 1; g.perN = 0;
    run_gemm(g, NB); }

  // c2: h2 = relu(bn2(conv3x3(h1)))
  { dim3 grid(HW / BN, CM / BM, NB);
    conv3_bf16_wmma<<<grid, 256, 0, stream>>>(w2b, h1, h2, bn2_g, bn2_b, inv_cb); }

  // rc: r = bn_r(Wr @ x)           M=C, N=HW, K=C
  { GemmArgs g = {};
    g.A = rcwb; g.sAm = C; g.sAk = 1; g.bA = 0;
    g.Bm = xb16; g.sBk = HW; g.sBn = 1; g.bB = (long long)C * HW;
    g.Cf = rxb; g.Cb = nullptr; g.sCm = HW; g.sCn = 1; g.bC = (long long)C * HW;
    g.resid = nullptr; g.scale = rbn_g; g.bias = rbn_b; g.bnInv = inv_cb;
    g.M = C; g.N = HW; g.Kd = C; g.act = 0; g.perN = 0;
    run_gemm(g, NB); }

  // c3 + residual: xb = relu(bn3(W3 @ h2) + r)   (in-place over r)
  { GemmArgs g = {};
    g.A = c3wb; g.sAm = CM; g.sAk = 1; g.bA = 0;
    g.Bm = h2; g.sBk = HW; g.sBn = 1; g.bB = (long long)CM * HW;
    g.Cf = rxb; g.Cb = xbb; g.sCm = HW; g.sCn = 1; g.bC = (long long)C * HW;
    g.resid = rxb; g.scale = bn3_g; g.bias = bn3_b; g.bnInv = inv_cb;
    g.M = C; g.N = HW; g.Kd = CM; g.act = 1; g.perN = 0;
    run_gemm(g, NB); }

  // lvc: e = relu(bn_lvc(Wl @ xb))
  { GemmArgs g = {};
    g.A = lvcwb; g.sAm = C; g.sAk = 1; g.bA = 0;
    g.Bm = xbb; g.sBk = HW; g.sBn = 1; g.bB = (long long)C * HW;
    g.Cf = nullptr; g.Cb = ebuf; g.sCm = HW; g.sCn = 1; g.bC = (long long)C * HW;
    g.resid = nullptr; g.scale = lvcbn_g; g.bias = lvcbn_b; g.bnInv = inv_lvc;
    g.M = C; g.N = HW; g.Kd = C; g.act = 1; g.perN = 0;
    run_gemm(g, NB); }

  x2_kernel<<<dim3(HW / 256, NB), 256, 0, stream>>>(ebuf, x2b);
  c2_kernel<<<1, KC, 0, stream>>>(cw, c2n);

  // xc = ef @ cw^T   M=HW(pixels), N=KC, K=C  (A = e transposed via strides)
  { GemmArgs g = {};
    g.A = ebuf; g.sAm = 1; g.sAk = HW; g.bA = (long long)C * HW;
    g.Bm = cwb; g.sBk = 1; g.sBn = C; g.bB = 0;
    g.Cf = xcb; g.Cb = nullptr; g.sCm = KC; g.sCn = 1; g.bC = (long long)HW * KC;
    g.resid = nullptr; g.scale = nullptr; g.bias = nullptr; g.bnInv = 1.f;
    g.M = HW; g.N = KC; g.Kd = C; g.act = 0; g.perN = 0;
    run_gemm(g, NB); }

  softmax_kernel<<<dim3(HW / 256, NB), 256, 0, stream>>>(xcb, x2b, c2n, escale, abuf);
  asum_kernel<<<(NB * KC + 255) / 256, 256, 0, stream>>>(abuf, asumb);

  // part1 = a^T @ ef   M=KC, N=C, K=HW
  { GemmArgs g = {};
    g.A = abuf; g.sAm = 1; g.sAk = KC; g.bA = (long long)HW * KC;
    g.Bm = ebuf; g.sBk = 1; g.sBn = HW; g.bB = (long long)C * HW;
    g.Cf = part1; g.Cb = nullptr; g.sCm = C; g.sCn = 1; g.bC = (long long)KC * C;
    g.resid = nullptr; g.scale = nullptr; g.bias = nullptr; g.bnInv = 1.f;
    g.M = KC; g.N = C; g.Kd = HW; g.act = 0; g.perN = 0;
    run_gemm(g, NB); }

  encpost_kernel<<<(NB * C + 255) / 256, 256, 0, stream>>>(
      part1, asumb, cw, enbn_g, enbn_b, inv_lvc, enb);

  // gam = sigmoid(en @ fc_w^T + fc_b)   M=NB, N=C, K=C  (per-N bias)
  { GemmArgs g = {};
    g.A = enb; g.sAm = C; g.sAk = 1; g.bA = 0;
    g.Bm = fcwb; g.sBk = 1; g.sBn = C; g.bB = 0;
    g.Cf = gam; g.Cb = nullptr; g.sCm = C; g.sCn = 1; g.bC = 0;
    g.resid = nullptr; g.scale = nullptr; g.bias = fc_b; g.bnInv = 1.f;
    g.M = NB; g.N = C; g.Kd = C; g.act = 2; g.perN = 1;
    run_gemm(g, 1); }

  final_kernel<<<(NB * C * HW + 255) / 256, 256, 0, stream>>>(rxb, gam, out);
}
